// Transformer_Model_78838419685979
// MI455X (gfx1250) — compile-verified, matrix operand order checked by reading, not run on hardware
//
#include <hip/hip_runtime.h>
#include <math.h>

typedef _Float16 v8h  __attribute__((ext_vector_type(8)));
typedef _Float16 v16h __attribute__((ext_vector_type(16)));
typedef float    v8f  __attribute__((ext_vector_type(8)));

// Problem dims (fixed by the reference)
constexpr int B_  = 128;
constexpr int T_  = 133;
constexpr int C_  = 512;
constexpr int H_  = 8;
constexpr int L_  = 8;
constexpr int V_  = 100;
constexpr int A_  = 10;
constexpr int FF_ = 4 * C_;        // 2048
constexpr int M_  = B_ * T_;       // 17024 = 133 * 128
constexpr int D_  = C_ / H_;       // 64

// GEMM tile config: 128x128 block tile, 8 waves in 4(M) x 2(N) grid,
// each wave owns a 32x64 sub-tile (2 A-frags x 4 B-frags = 8 WMMAs).
constexpr int TM = 128;
constexpr int TN = 128;
constexpr int KS = 32;             // one wmma k-step

// ---------------------------------------------------------------------------
// Embedding: x[b,t,:] = tok + pos + (admet @ W + b) + scaffold
// ---------------------------------------------------------------------------
__global__ __launch_bounds__(256) void embed_kernel(
    const int* __restrict__ idx, const float* __restrict__ props,
    const int* __restrict__ scaf, const float* __restrict__ tok_e,
    const float* __restrict__ pos_e, const float* __restrict__ adW,
    const float* __restrict__ adb, const float* __restrict__ scaf_e,
    float* __restrict__ x)
{
    int bt = blockIdx.x;                 // 0..M-1
    int b  = bt / T_;
    int t  = bt % T_;
    int tk = idx[bt];
    long sc = scaf[b];
    float p[A_];
#pragma unroll
    for (int a = 0; a < A_; ++a) p[a] = props[b * A_ + a];
    for (int c = threadIdx.x; c < C_; c += blockDim.x) {
        float acc = tok_e[(long)tk * C_ + c] + pos_e[(long)t * C_ + c]
                  + adb[c] + scaf_e[sc * C_ + c];
#pragma unroll
        for (int a = 0; a < A_; ++a) acc += p[a] * adW[a * C_ + c];
        x[(long)bt * C_ + c] = acc;
    }
}

// ---------------------------------------------------------------------------
// Weight convert+transpose: WT[n][k] = f16(W[k][n]); rows n >= N zero-padded.
// ---------------------------------------------------------------------------
__global__ __launch_bounds__(256) void convT_kernel(
    const float* __restrict__ W, _Float16* __restrict__ WT,
    int K, int N, long total)
{
    long i = (long)blockIdx.x * blockDim.x + threadIdx.x;
    if (i >= total) return;
    int n = (int)(i / K);
    int k = (int)(i % K);
    WT[i] = (n < N) ? (_Float16)W[(long)k * N + n] : (_Float16)0.f;
}

// ---------------------------------------------------------------------------
// LayerNorm over C=512: one wave32 per row, f16 output (WMMA A operand).
// ---------------------------------------------------------------------------
__global__ __launch_bounds__(256) void ln_kernel(
    const float* __restrict__ x, const float* __restrict__ w,
    const float* __restrict__ b, _Float16* __restrict__ out)
{
    int row  = blockIdx.x * 8 + (threadIdx.x >> 5);
    int lane = threadIdx.x & 31;
    const float* xr = x + (long)row * C_;
    float s = 0.f;
    for (int i = lane; i < C_; i += 32) s += xr[i];
#pragma unroll
    for (int o = 16; o; o >>= 1) s += __shfl_xor(s, o, 32);
    float mean = s * (1.f / C_);
    float vs = 0.f;
    for (int i = lane; i < C_; i += 32) { float d = xr[i] - mean; vs += d * d; }
#pragma unroll
    for (int o = 16; o; o >>= 1) vs += __shfl_xor(vs, o, 32);
    float rstd = rsqrtf(vs * (1.f / C_) + 1e-5f);
    for (int i = lane; i < C_; i += 32)
        out[(long)row * C_ + i] = (_Float16)((xr[i] - mean) * rstd * w[i] + b[i]);
}

// ---------------------------------------------------------------------------
// WMMA GEMM: Out[M,N] = A16[M,K] * WT16[N,K]^T (+bias) (+epilogue)
//   MODE 0: store f16                          (Q/K/V projections, no bias)
//   MODE 1: +bias, exact GELU, store f16       (FFN up-proj)
//   MODE 2: +bias, accumulate into f32 resid   (attn out-proj, FFN down-proj)
//   MODE 3: +bias, store f32 logits, n < nvalid guard, row stride nvalid
// Block: 256 threads = 8 waves (4 in M x 2 in N); tile 128x128; K step 32.
// Per wave per K-step: 4 ds_load_b128 (A) + 8 (B) feeding 8 v_wmma.
// ---------------------------------------------------------------------------
template <int MODE>
__global__ __launch_bounds__(256) void gemm_wmma(
    const _Float16* __restrict__ Amat, const _Float16* __restrict__ WT,
    const float* __restrict__ bias, _Float16* __restrict__ outH,
    float* __restrict__ outF, int N, int K, int nvalid)
{
    __shared__ _Float16 sA[TM * KS];   // [row][k]  8 KB
    __shared__ _Float16 sB[TN * KS];   // [col][k]  8 KB

    const int m0   = blockIdx.x * TM;
    const int n0   = blockIdx.y * TN;
    const int tid  = threadIdx.x;
    const int wave = tid >> 5;
    const int lane = tid & 31;
    const int wm   = wave & 3;         // M group: 32 rows each
    const int wn   = wave >> 2;        // N group: 64 cols each

    v8f acc[2][4] = {};

    // per-lane fragment offsets within a [16 x 32] tile:
    // lanes 0-15 -> row/col = lane, K chunks {0..7,16..23};
    // lanes 16-31 -> K chunks {8..15,24..31}.
    const int frow = lane & 15;
    const int hi8  = (lane >> 4) << 3;

    for (int k0 = 0; k0 < K; k0 += KS) {
        // stage A tile: 128x32 halves, 16 halves/thread (2 x b128)
#pragma unroll
        for (int i = 0; i < 2; ++i) {
            int c   = tid + i * 256;           // 0..511
            int row = c >> 2;
            int off = (c & 3) << 3;
            *(v8h*)&sA[row * KS + off] =
                *(const v8h*)&Amat[(long)(m0 + row) * K + k0 + off];
        }
        // stage B tile: 128x32 halves, same pattern
#pragma unroll
        for (int i = 0; i < 2; ++i) {
            int c   = tid + i * 256;
            int n   = c >> 2;
            int off = (c & 3) << 3;
            *(v8h*)&sB[n * KS + off] =
                *(const v8h*)&WT[(long)(n0 + n) * K + k0 + off];
        }
        __syncthreads();

        // two A fragments for this wave's 32-row strip
        const _Float16* ap0 = &sA[((wm * 32 + frow) * KS) + hi8];
        const _Float16* ap1 = ap0 + 16 * KS;
        v8h a0lo = *(const v8h*)ap0;
        v8h a0hi = *(const v8h*)(ap0 + 16);
        v8h a1lo = *(const v8h*)ap1;
        v8h a1hi = *(const v8h*)(ap1 + 16);
        v16h af0 = __builtin_shufflevector(a0lo, a0hi,
            0,1,2,3,4,5,6,7,8,9,10,11,12,13,14,15);
        v16h af1 = __builtin_shufflevector(a1lo, a1hi,
            0,1,2,3,4,5,6,7,8,9,10,11,12,13,14,15);

#pragma unroll
        for (int nt = 0; nt < 4; ++nt) {
            const _Float16* bp = &sB[((wn * 64 + nt * 16 + frow) * KS) + hi8];
            v8h blo = *(const v8h*)bp;
            v8h bhi = *(const v8h*)(bp + 16);
            v16h bf = __builtin_shufflevector(blo, bhi,
                0,1,2,3,4,5,6,7,8,9,10,11,12,13,14,15);
            acc[0][nt] = __builtin_amdgcn_wmma_f32_16x16x32_f16(
                false, af0, false, bf, (short)0, acc[0][nt], false, false);
            acc[1][nt] = __builtin_amdgcn_wmma_f32_16x16x32_f16(
                false, af1, false, bf, (short)0, acc[1][nt], false, false);
        }
        __syncthreads();
    }

    // Epilogue. C/D layout: VGPR r -> M = r (+8 for lanes 16-31), N = lane&15.
    const int nl = lane & 15;
#pragma unroll
    for (int a = 0; a < 2; ++a) {
        const int mbase = m0 + wm * 32 + a * 16 + ((lane >> 4) << 3);
#pragma unroll
        for (int nt = 0; nt < 4; ++nt) {
            int gn = n0 + wn * 64 + nt * 16 + nl;
            if constexpr (MODE == 3) {
                if (gn >= nvalid) continue;
                float bv = bias[gn];
#pragma unroll
                for (int r = 0; r < 8; ++r)
                    outF[(long)(mbase + r) * nvalid + gn] = acc[a][nt][r] + bv;
            } else if constexpr (MODE == 2) {
                float bv = bias[gn];
#pragma unroll
                for (int r = 0; r < 8; ++r)
                    outF[(long)(mbase + r) * N + gn] += acc[a][nt][r] + bv;
            } else if constexpr (MODE == 1) {
                float bv = bias[gn];
#pragma unroll
                for (int r = 0; r < 8; ++r) {
                    float v = acc[a][nt][r] + bv;
                    v = 0.5f * v * (1.0f + erff(v * 0.70710678118654752f));
                    outH[(long)(mbase + r) * N + gn] = (_Float16)v;
                }
            } else {
#pragma unroll
                for (int r = 0; r < 8; ++r)
                    outH[(long)(mbase + r) * N + gn] = (_Float16)acc[a][nt][r];
            }
        }
    }
}

// ---------------------------------------------------------------------------
// Causal attention, one wave32 per (b, h, t). Lanes split d=64 (2 elems/lane).
// q/k/v are f16 [M][C] with head offset h*64; y written f16 (WMMA A operand).
// ---------------------------------------------------------------------------
__global__ __launch_bounds__(256) void attn_kernel(
    const _Float16* __restrict__ q, const _Float16* __restrict__ k,
    const _Float16* __restrict__ v, _Float16* __restrict__ y)
{
    __shared__ float sprob[8][T_ + 27];    // per-wave score buffer (160 floats)
    const int wid  = threadIdx.x >> 5;
    const int lane = threadIdx.x & 31;
    const int gw   = blockIdx.x * 8 + wid;       // 0 .. B*H*T-1 (exact)
    const int t  = gw % T_;
    const int bh = gw / T_;
    const int h  = bh % H_;
    const int b  = bh / H_;
    const long base = (long)b * T_ * C_ + (long)h * D_;

    const _Float16* qr = q + base + (long)t * C_;
    float q0 = (float)qr[lane], q1 = (float)qr[lane + 32];
    float* prob = sprob[wid];

    float mx = -1e30f;
    for (int s = 0; s <= t; ++s) {
        const _Float16* kr = k + base + (long)s * C_;
        float p = q0 * (float)kr[lane] + q1 * (float)kr[lane + 32];
#pragma unroll
        for (int o = 16; o; o >>= 1) p += __shfl_xor(p, o, 32);
        p *= 0.125f;                       // 1/sqrt(64)
        if (lane == 0) prob[s] = p;
        mx = fmaxf(mx, p);
    }
    float den = 0.f;
    for (int s = lane; s <= t; s += 32) {
        float e = __expf(prob[s] - mx);
        prob[s] = e;
        den += e;
    }
#pragma unroll
    for (int o = 16; o; o >>= 1) den += __shfl_xor(den, o, 32);
    float inv = 1.f / den;

    float a0 = 0.f, a1 = 0.f;
    for (int s = 0; s <= t; ++s) {
        float p = prob[s] * inv;
        const _Float16* vr = v + base + (long)s * C_;
        a0 += p * (float)vr[lane];
        a1 += p * (float)vr[lane + 32];
    }
    _Float16* yr = y + base + (long)t * C_;
    yr[lane]      = (_Float16)a0;
    yr[lane + 32] = (_Float16)a1;
}

// ---------------------------------------------------------------------------
// Host orchestration
// ---------------------------------------------------------------------------
extern "C" void kernel_launch(void* const* d_in, const int* in_sizes, int n_in,
                              void* d_out, int out_size, void* d_ws, size_t ws_size,
                              hipStream_t stream)
{
    const int*   idx     = (const int*)  d_in[0];
    const float* props   = (const float*)d_in[1];
    const int*   scaf    = (const int*)  d_in[2];
    const float* tok_e   = (const float*)d_in[3];
    const float* pos_e   = (const float*)d_in[4];
    const float* adW     = (const float*)d_in[5];
    const float* adb     = (const float*)d_in[6];
    const float* scaf_e  = (const float*)d_in[7];
    const float* ln1w    = (const float*)d_in[8];
    const float* ln1b    = (const float*)d_in[9];
    const float* Wq      = (const float*)d_in[10];
    const float* Wk      = (const float*)d_in[11];
    const float* Wv      = (const float*)d_in[12];
    const float* Wp      = (const float*)d_in[13];
    const float* bp      = (const float*)d_in[14];
    const float* ln2w    = (const float*)d_in[15];
    const float* ln2b    = (const float*)d_in[16];
    const float* W1      = (const float*)d_in[17];
    const float* b1      = (const float*)d_in[18];
    const float* W2      = (const float*)d_in[19];
    const float* b2      = (const float*)d_in[20];
    const float* lnfw    = (const float*)d_in[21];
    const float* lnfb    = (const float*)d_in[22];
    const float* Wout    = (const float*)d_in[23];
    const float* bout    = (const float*)d_in[24];
    float* logits = (float*)d_out;

    // -------- workspace layout --------
    char* w = (char*)d_ws;
    size_t off = 0;
    auto take = [&](size_t bytes) { char* p = w + off; off += (bytes + 255) & ~size_t(255); return p; };
    float*    x   = (float*)   take((size_t)M_ * C_  * 4);
    _Float16* hbf = (_Float16*)take((size_t)M_ * C_  * 2);
    _Float16* qb  = (_Float16*)take((size_t)M_ * C_  * 2);
    _Float16* kb  = (_Float16*)take((size_t)M_ * C_  * 2);
    _Float16* vb  = (_Float16*)take((size_t)M_ * C_  * 2);
    _Float16* yb  = (_Float16*)take((size_t)M_ * C_  * 2);
    _Float16* gb  = (_Float16*)take((size_t)M_ * FF_ * 2);
    const size_t perLayerH = (size_t)4 * C_ * C_ + 2 * (size_t)C_ * FF_;
    _Float16* wts = (_Float16*)take(perLayerH * L_ * 2);
    constexpr int VP = 128;                      // padded logits width
    _Float16* woT = (_Float16*)take((size_t)VP * C_ * 2);
    (void)ws_size; (void)n_in; (void)in_sizes; (void)out_size;

    auto conv = [&](const float* src, _Float16* dst, int K, int N, int Npad) {
        long total = (long)Npad * K;
        convT_kernel<<<(int)((total + 255) / 256), 256, 0, stream>>>(src, dst, K, N, total);
    };

    // -------- convert + transpose all weights to f16 [N][K] --------
    for (int l = 0; l < L_; ++l) {
        _Float16* base = wts + (size_t)l * perLayerH;
        _Float16* wqT = base;
        _Float16* wkT = wqT + (size_t)C_ * C_;
        _Float16* wvT = wkT + (size_t)C_ * C_;
        _Float16* wpT = wvT + (size_t)C_ * C_;
        _Float16* w1T = wpT + (size_t)C_ * C_;   // [FF][C]
        _Float16* w2T = w1T + (size_t)FF_ * C_;  // [C][FF]
        conv(Wq + (size_t)l * C_ * C_,  wqT, C_,  C_,  C_);
        conv(Wk + (size_t)l * C_ * C_,  wkT, C_,  C_,  C_);
        conv(Wv + (size_t)l * C_ * C_,  wvT, C_,  C_,  C_);
        conv(Wp + (size_t)l * C_ * C_,  wpT, C_,  C_,  C_);
        conv(W1 + (size_t)l * C_ * FF_, w1T, C_,  FF_, FF_);
        conv(W2 + (size_t)l * FF_ * C_, w2T, FF_, C_,  C_);
    }
    conv(Wout, woT, C_, V_, VP);                 // zero-padded to 128 cols

    // -------- embeddings --------
    embed_kernel<<<M_, 256, 0, stream>>>(idx, props, scaf, tok_e, pos_e, adW, adb, scaf_e, x);

    const dim3 gC(M_ / TM, C_  / TN);            // 133 x 4
    const dim3 gF(M_ / TM, FF_ / TN);            // 133 x 16
    const dim3 gV(M_ / TM, VP  / TN);            // 133 x 1
    const int lnBlocks = M_ / 8;                 // 2128
    const int attnBlocks = (B_ * H_ * T_) / 8;   // 17024

    for (int l = 0; l < L_; ++l) {
        _Float16* base = wts + (size_t)l * perLayerH;
        _Float16* wqT = base;
        _Float16* wkT = wqT + (size_t)C_ * C_;
        _Float16* wvT = wkT + (size_t)C_ * C_;
        _Float16* wpT = wvT + (size_t)C_ * C_;
        _Float16* w1T = wpT + (size_t)C_ * C_;
        _Float16* w2T = w1T + (size_t)FF_ * C_;

        ln_kernel<<<lnBlocks, 256, 0, stream>>>(x, ln1w + l * C_, ln1b + l * C_, hbf);
        gemm_wmma<0><<<gC, 256, 0, stream>>>(hbf, wqT, nullptr, qb, nullptr, C_, C_, 0);
        gemm_wmma<0><<<gC, 256, 0, stream>>>(hbf, wkT, nullptr, kb, nullptr, C_, C_, 0);
        gemm_wmma<0><<<gC, 256, 0, stream>>>(hbf, wvT, nullptr, vb, nullptr, C_, C_, 0);
        attn_kernel<<<attnBlocks, 256, 0, stream>>>(qb, kb, vb, yb);
        gemm_wmma<2><<<gC, 256, 0, stream>>>(yb, wpT, bp + l * C_, nullptr, x, C_, C_, 0);
        ln_kernel<<<lnBlocks, 256, 0, stream>>>(x, ln2w + l * C_, ln2b + l * C_, hbf);
        gemm_wmma<1><<<gF, 256, 0, stream>>>(hbf, w1T, b1 + l * FF_, gb, nullptr, FF_, C_, 0);
        gemm_wmma<2><<<gC, 256, 0, stream>>>(gb, w2T, b2 + l * C_, nullptr, x, C_, FF_, 0);
    }

    ln_kernel<<<lnBlocks, 256, 0, stream>>>(x, lnfw, lnfb, hbf);
    gemm_wmma<3><<<gV, 256, 0, stream>>>(hbf, woT, bout, nullptr, logits, VP, C_, V_);
}
